// AdvancedPhotonicTransformer_77068893160348
// MI455X (gfx1250) — matrix operand order repaired and checked by reading, not compiled, not run on hardware
//
#include <hip/hip_runtime.h>
#include <hip/hip_bf16.h>
#include <stdint.h>

#define S_LEN   512
#define D_DIM   512
#define NHEAD   8
#define HEADD   64
#define NLAYER  6
#define NWL     16
#define MROWS   1024           // B*S
#define KERR_C  2.7e-12f       // 2.7e-18 * 1e6
#define LN_EPS_C 1e-5f
#define LDT     72             // LDS tile row stride (elements); 144B = 9*16B

typedef __attribute__((ext_vector_type(16))) __bf16 v16bf;
typedef __attribute__((ext_vector_type(8)))  __bf16 bf16x8;
typedef __attribute__((ext_vector_type(8)))  float  v8f;
typedef __attribute__((ext_vector_type(4)))  int    v4i;

// ---------------------------------------------------------------------------
// 16-byte tile-staging copy: global bf16 -> LDS bf16 via gfx1250 async copy
// (ASYNCcnt path). Probe-learned signature: (AS1 v4i*, AS3 v4i*, i32, i32).
// ---------------------------------------------------------------------------
#if __has_builtin(__builtin_amdgcn_global_load_async_to_lds_b128)
#define PT_ASYNC 1
#else
#define PT_ASYNC 0
#endif

typedef __attribute__((address_space(1))) v4i* glb4_t;
typedef __attribute__((address_space(3))) v4i* lds4_t;

__device__ __forceinline__ void cp16(__bf16* dst_lds, const __bf16* src_glb) {
#if PT_ASYNC
  __builtin_amdgcn_global_load_async_to_lds_b128(
      (glb4_t)(uintptr_t)src_glb,
      (lds4_t)(uint32_t)(uintptr_t)dst_lds,
      0, 0);
#else
  *(bf16x8*)dst_lds = *(const bf16x8*)src_glb;
#endif
}

__device__ __forceinline__ void stage_fence() {
#if PT_ASYNC
#if __has_builtin(__builtin_amdgcn_s_wait_asynccnt)
  __builtin_amdgcn_s_wait_asynccnt(0);
#else
  asm volatile("s_wait_asynccnt 0" ::: "memory");
#endif
#endif
  __syncthreads();
}

// Stage a 64x64 bf16 tile (global row stride gld elements) into LDS [64][LDT].
// 512 16B-chunks, 128 threads -> exactly 4 unconditional copies per thread.
__device__ __forceinline__ void stage_tile(__bf16* lds, const __bf16* gsrc,
                                           int gld, int tid) {
#pragma unroll
  for (int j = 0; j < 4; ++j) {
    const int i = tid + j * 128;
    const int r = i >> 3, c = (i & 7) * 8;
    cp16(lds + r * LDT + c, gsrc + (size_t)r * gld + c);
  }
}

// ---------------------------------------------------------------------------
// WMMA fragment loaders. Both LDS tiles are [row][K] with contiguous K, so
// every fragment is two 16B contiguous LDS loads (ds_load_b128).
// A 16x32: lane<16 -> K {0..7,16..23}; lane>=16 -> K {8..15,24..31}
// B 32x16: lane<16 -> K {0..7,16..23}; lane>=16 -> K {8..15,24..31}; N = lane&15
// ---------------------------------------------------------------------------
__device__ __forceinline__ v16bf frag_a(const __bf16* Ts, int row0, int lane) {
  const __bf16* p = Ts + (row0 + (lane & 15)) * LDT + ((lane >> 4) << 3);
  bf16x8 lo = *(const bf16x8*)p;
  bf16x8 hi = *(const bf16x8*)(p + 16);
  return __builtin_shufflevector(lo, hi, 0,1,2,3,4,5,6,7,8,9,10,11,12,13,14,15);
}

__device__ __forceinline__ v16bf frag_b(const __bf16* Ts, int col0, int lane) {
  const __bf16* p = Ts + (col0 + (lane & 15)) * LDT + ((lane >> 4) << 3);
  bf16x8 lo = *(const bf16x8*)p;
  bf16x8 hi = *(const bf16x8*)(p + 16);
  return __builtin_shufflevector(lo, hi, 0,1,2,3,4,5,6,7,8,9,10,11,12,13,14,15);
}

#define DECL_ACC(acc)                                   \
  const v8f z_ = {0.f,0.f,0.f,0.f,0.f,0.f,0.f,0.f};     \
  v8f acc[2][2];                                        \
  _Pragma("unroll") for (int i_ = 0; i_ < 2; ++i_)      \
  _Pragma("unroll") for (int j_ = 0; j_ < 2; ++j_) acc[i_][j_] = z_;

// Two K-slabs (ks = 0, 32) of 2x2 WMMA tiles over the staged 64x64 tiles.
#define WMMA_TILE(acc, As, Bs, wm, wn, lane)                                  \
  _Pragma("unroll") for (int ks_ = 0; ks_ < 64; ks_ += 32) {                  \
    _Pragma("unroll") for (int tm_ = 0; tm_ < 2; ++tm_) {                     \
      v16bf a_ = frag_a((As) + ks_, (wm) + tm_ * 16, lane);                   \
      _Pragma("unroll") for (int tn_ = 0; tn_ < 2; ++tn_) {                   \
        v16bf b_ = frag_b((Bs) + ks_, (wn) + tn_ * 16, lane);                 \
        acc[tm_][tn_] = __builtin_amdgcn_wmma_f32_16x16x32_bf16(              \
            false, a_, false, b_, (short)0, acc[tm_][tn_], false, false);     \
      }                                                                       \
    }                                                                         \
  }

// ---------------------------------------------------------------------------
// GEMM: Y[M,N] = epilogue( A[M,K](bf16) @ Bt[N,K](bf16)^T + bias )
// act==1: tanh(2v)/(1+|v|); R: fp32 residual; obf: write bf16 else fp32.
// 128 threads = 4 waves, 64x64 macro tile, 32x32 per wave, K-step 64.
// ---------------------------------------------------------------------------
__global__ __launch_bounds__(128)
void pt_gemm(const __bf16* __restrict__ A, const __bf16* __restrict__ Bt,
             const float* __restrict__ bias, const float* __restrict__ R,
             void* __restrict__ Yv, int M, int N, int K, int act, int obf)
{
  __shared__ __align__(16) __bf16 As[64 * LDT];
  __shared__ __align__(16) __bf16 Bs[64 * LDT];
  const int tid  = threadIdx.x;
  const int lane = tid & 31;
  const int wave = tid >> 5;
  const int m0 = blockIdx.y * 64;
  const int n0 = blockIdx.x * 64;
  const int wm = (wave >> 1) * 32;
  const int wn = (wave & 1) * 32;
  const __bf16* a0 = A  + (size_t)m0 * K;
  const __bf16* b0 = Bt + (size_t)n0 * K;

  DECL_ACC(acc);
  for (int k0 = 0; k0 < K; k0 += 64) {
    stage_tile(As, a0 + k0, K, tid);
    stage_tile(Bs, b0 + k0, K, tid);
    if (k0 + 64 < K) {  // speculative prefetch of next K chunk
      __builtin_prefetch(a0 + (size_t)(tid >> 1) * K + k0 + 64 + (tid & 1) * 32, 0, 1);
      __builtin_prefetch(b0 + (size_t)(tid >> 1) * K + k0 + 64 + (tid & 1) * 32, 0, 1);
    }
    stage_fence();
    WMMA_TILE(acc, As, Bs, wm, wn, lane);
    __syncthreads();
  }

#pragma unroll
  for (int tm = 0; tm < 2; ++tm)
#pragma unroll
    for (int tn = 0; tn < 2; ++tn)
#pragma unroll
      for (int r = 0; r < 8; ++r) {
        int row = m0 + wm + tm * 16 + ((lane >> 4) << 3) + r;
        int col = n0 + wn + tn * 16 + (lane & 15);
        float val = acc[tm][tn][r] + bias[col];
        if (act == 1) val = tanhf(2.f * val) / (1.f + fabsf(val));
        if (R) val += R[(size_t)row * N + col];
        if (obf) ((__bf16*)Yv)[(size_t)row * N + col] = (__bf16)val;
        else     ((float*) Yv)[(size_t)row * N + col] = val;
      }
}

// ---------------------------------------------------------------------------
// Scores: SC[bh,s,t] = (1/8) * sum_hd q[b,s,h,hd] * k[b,t,h,hd]
// B operand tile [t][hd] is exactly k's natural row-major layout. K=64: one step.
// grid = (8, 8, 16)
// ---------------------------------------------------------------------------
__global__ __launch_bounds__(128)
void pt_scores(const __bf16* __restrict__ Q, const __bf16* __restrict__ Kx,
               float* __restrict__ SC)
{
  __shared__ __align__(16) __bf16 As[64 * LDT];
  __shared__ __align__(16) __bf16 Bs[64 * LDT];
  const int tid  = threadIdx.x;
  const int lane = tid & 31;
  const int wave = tid >> 5;
  const int n0 = blockIdx.x * 64;
  const int m0 = blockIdx.y * 64;
  const int bh = blockIdx.z, b = bh >> 3, h = bh & 7;
  const int wm = (wave >> 1) * 32;
  const int wn = (wave & 1) * 32;
  const __bf16* qb = Q  + ((size_t)b * S_LEN + m0) * D_DIM + h * HEADD;
  const __bf16* kb = Kx + ((size_t)b * S_LEN + n0) * D_DIM + h * HEADD;

  DECL_ACC(acc);
  stage_tile(As, qb, D_DIM, tid);
  stage_tile(Bs, kb, D_DIM, tid);
  stage_fence();
  WMMA_TILE(acc, As, Bs, wm, wn, lane);

  float* out = SC + (size_t)bh * S_LEN * S_LEN;
#pragma unroll
  for (int tm = 0; tm < 2; ++tm)
#pragma unroll
    for (int tn = 0; tn < 2; ++tn)
#pragma unroll
      for (int r = 0; r < 8; ++r) {
        int row = m0 + wm + tm * 16 + ((lane >> 4) << 3) + r;
        int col = n0 + wn + tn * 16 + (lane & 15);
        out[(size_t)row * S_LEN + col] = 0.125f * acc[tm][tn][r];
      }
}

// ---------------------------------------------------------------------------
// A @ V per wavelength: attn[b,s,h*64+hd] (+)= (iw[wl]/16) * P[bh,s,:] @ V
// B operand from Vt[b][d][t] (pre-transposed), tile [hd][t]. grid = (1,8,16)
// ---------------------------------------------------------------------------
__global__ __launch_bounds__(128)
void pt_av(const __bf16* __restrict__ P, const __bf16* __restrict__ Vt,
           float* __restrict__ A, const float* __restrict__ iw, int wl, int accum)
{
  __shared__ __align__(16) __bf16 As[64 * LDT];
  __shared__ __align__(16) __bf16 Bs[64 * LDT];
  const int tid  = threadIdx.x;
  const int lane = tid & 31;
  const int wave = tid >> 5;
  const int m0 = blockIdx.y * 64;
  const int bh = blockIdx.z, b = bh >> 3, h = bh & 7;
  const int wm = (wave >> 1) * 32;
  const int wn = (wave & 1) * 32;
  const __bf16* pb = P  + ((size_t)bh * S_LEN + m0) * S_LEN;
  const __bf16* vb = Vt + ((size_t)b * D_DIM + h * HEADD) * S_LEN;

  DECL_ACC(acc);
  for (int k0 = 0; k0 < S_LEN; k0 += 64) {
    stage_tile(As, pb + k0, S_LEN, tid);
    stage_tile(Bs, vb + k0, S_LEN, tid);
    stage_fence();
    WMMA_TILE(acc, As, Bs, wm, wn, lane);
    __syncthreads();
  }

  const float scale = iw[wl] * (1.0f / (float)NWL);
#pragma unroll
  for (int tm = 0; tm < 2; ++tm)
#pragma unroll
    for (int tn = 0; tn < 2; ++tn)
#pragma unroll
      for (int r = 0; r < 8; ++r) {
        int row = m0 + wm + tm * 16 + ((lane >> 4) << 3) + r;   // s
        int col = wn + tn * 16 + (lane & 15);                   // hd
        size_t idx = ((size_t)b * S_LEN + row) * D_DIM + h * HEADD + col;
        float val = scale * acc[tm][tn][r];
        if (accum) val += A[idx];
        A[idx] = val;
      }
}

// ---------------------------------------------------------------------------
// Weight transpose+convert: fp32 W[K][N] -> bf16 Wt[N][K]. grid = (N/64, K/64)
// ---------------------------------------------------------------------------
__global__ __launch_bounds__(256)
void pt_transpose_w(const float* __restrict__ W, __bf16* __restrict__ Wt,
                    int K, int N)
{
  __shared__ float T[64][65];
  const int k0 = blockIdx.y * 64, n0 = blockIdx.x * 64;
#pragma unroll
  for (int i = threadIdx.x; i < 4096; i += 256) {
    int r = i >> 6, c = i & 63;
    T[r][c] = W[(size_t)(k0 + r) * N + n0 + c];
  }
  __syncthreads();
#pragma unroll
  for (int i = threadIdx.x; i < 4096; i += 256) {
    int r = i >> 6, c = i & 63;
    Wt[(size_t)(n0 + r) * K + k0 + c] = (__bf16)T[c][r];
  }
}

// bf16 batched transpose: V[b][S][D] -> Vt[b][D][S]. grid = (8, 8, 2)
__global__ __launch_bounds__(256)
void pt_transpose_v(const __bf16* __restrict__ V, __bf16* __restrict__ Vt)
{
  __shared__ __bf16 T[64][72];
  const int b = blockIdx.z, t0 = blockIdx.y * 64, d0 = blockIdx.x * 64;
  const size_t base = (size_t)b * S_LEN * D_DIM;
#pragma unroll
  for (int i = threadIdx.x; i < 4096; i += 256) {
    int r = i >> 6, c = i & 63;
    T[r][c] = V[base + (size_t)(t0 + r) * D_DIM + d0 + c];
  }
  __syncthreads();
#pragma unroll
  for (int i = threadIdx.x; i < 4096; i += 256) {
    int r = i >> 6, c = i & 63;
    Vt[base + (size_t)(d0 + r) * S_LEN + t0 + c] = T[c][r];
  }
}

// fp32 -> bf16 pack (grid covers count/256)
__global__ __launch_bounds__(256)
void pt_pack(const float* __restrict__ X, __bf16* __restrict__ Y)
{
  const int i = blockIdx.x * 256 + threadIdx.x;
  Y[i] = (__bf16)X[i];
}

// ---------------------------------------------------------------------------
// Embedding gather + positional add.
// ---------------------------------------------------------------------------
__global__ __launch_bounds__(256)
void pt_embed(const int* __restrict__ ids, const float* __restrict__ tok,
              const float* __restrict__ pos, float* __restrict__ X)
{
  const int i = blockIdx.x * 256 + threadIdx.x;
  const int m = i >> 9, d = i & 511;
  X[i] = tok[(size_t)ids[m] * D_DIM + d] + pos[(size_t)(m & 511) * D_DIM + d];
}

// ---------------------------------------------------------------------------
// LayerNorm, bf16 output (feeds GEMM A operand). One 256-thread block per row.
// ---------------------------------------------------------------------------
__global__ __launch_bounds__(256)
void pt_layernorm(const float* __restrict__ X, const float* __restrict__ g,
                  const float* __restrict__ bta, __bf16* __restrict__ Y)
{
  __shared__ float red[256];
  const int tid = threadIdx.x;
  const size_t row = blockIdx.x;
  const float* xr = X + row * D_DIM;
  float v0 = xr[tid], v1 = xr[tid + 256];

  red[tid] = v0 + v1; __syncthreads();
  for (int o = 128; o > 0; o >>= 1) { if (tid < o) red[tid] += red[tid + o]; __syncthreads(); }
  const float mu = red[0] * (1.0f / D_DIM); __syncthreads();

  const float d0 = v0 - mu, d1 = v1 - mu;
  red[tid] = d0 * d0 + d1 * d1; __syncthreads();
  for (int o = 128; o > 0; o >>= 1) { if (tid < o) red[tid] += red[tid + o]; __syncthreads(); }
  const float inv = rsqrtf(red[0] * (1.0f / D_DIM) + LN_EPS_C);

  __bf16* yr = Y + row * D_DIM;
  yr[tid]       = (__bf16)(d0 * inv * g[tid]       + bta[tid]);
  yr[tid + 256] = (__bf16)(d1 * inv * g[tid + 256] + bta[tid + 256]);
}

// ---------------------------------------------------------------------------
// Per-wavelength fused scale + Kerr + softmax -> bf16 P. grid = 16*512
// ---------------------------------------------------------------------------
__global__ __launch_bounds__(256)
void pt_softmax(const float* __restrict__ SC, const float* __restrict__ phase,
                __bf16* __restrict__ P, int wl)
{
  __shared__ float red[256];
  const int tid = threadIdx.x;
  const int rowid = blockIdx.x;
  const int bh = rowid >> 9, s = rowid & 511, h = bh & 7;
  const float c  = cosf(phase[wl * NHEAD + h]);
  const float c2 = c * c;
  const float* base = SC + (size_t)bh * S_LEN * S_LEN + (size_t)s * S_LEN;

  float v0 = base[tid] * c2;
  float v1 = base[tid + 256] * c2;
  v0 = v0 * cosf(KERR_C * v0 * v0);
  v1 = v1 * cosf(KERR_C * v1 * v1);

  red[tid] = fmaxf(v0, v1); __syncthreads();
  for (int o = 128; o > 0; o >>= 1) { if (tid < o) red[tid] = fmaxf(red[tid], red[tid + o]); __syncthreads(); }
  const float mx = red[0]; __syncthreads();

  const float e0 = __expf(v0 - mx), e1 = __expf(v1 - mx);
  red[tid] = e0 + e1; __syncthreads();
  for (int o = 128; o > 0; o >>= 1) { if (tid < o) red[tid] += red[tid + o]; __syncthreads(); }
  const float inv = 1.0f / red[0];

  __bf16* out = P + (size_t)bh * S_LEN * S_LEN + (size_t)s * S_LEN;
  out[tid]       = (__bf16)(e0 * inv);
  out[tid + 256] = (__bf16)(e1 * inv);
}

// ---------------------------------------------------------------------------
// Host orchestration.
// ---------------------------------------------------------------------------
extern "C" void kernel_launch(void* const* d_in, const int* in_sizes, int n_in,
                              void* d_out, int out_size, void* d_ws, size_t ws_size,
                              hipStream_t stream)
{
  (void)in_sizes; (void)n_in; (void)out_size; (void)ws_size;
  const int*   ids = (const int*)  d_in[0];
  const float* tok = (const float*)d_in[1];
  const float* pos = (const float*)d_in[2];
  const float* ow_final = (const float*)d_in[3 + NLAYER * 18];
  const float* ob_final = (const float*)d_in[3 + NLAYER * 18 + 1];

  char* wsb = (char*)d_ws;
  const size_t MB = 1u << 20;
  float*  x     = (float*) (wsb);             // 2 MB  [M][D] fp32 residual
  float*  attn  = (float*) (wsb +  2 * MB);   // 2 MB  [M][D] fp32 accum
  float*  sc    = (float*) (wsb +  4 * MB);   // 16 MB [16][S][S] fp32
  __bf16* xnb   = (__bf16*)(wsb + 20 * MB);   // 1 MB  [M][D]
  __bf16* q     = (__bf16*)(wsb + 21 * MB);   // 1 MB
  __bf16* k     = (__bf16*)(wsb + 22 * MB);   // 1 MB
  __bf16* v     = (__bf16*)(wsb + 23 * MB);   // 1 MB
  __bf16* vt    = (__bf16*)(wsb + 24 * MB);   // 1 MB  [b][D][S]
  __bf16* attnb = (__bf16*)(wsb + 25 * MB);   // 1 MB
  __bf16* ffnb  = (__bf16*)(wsb + 26 * MB);   // 4 MB  [M][2048]
  __bf16* xb    = (__bf16*)(wsb + 30 * MB);   // 1 MB
  __bf16* P     = (__bf16*)(wsb + 31 * MB);   // 8 MB  [16][S][S]
  __bf16* wt    = (__bf16*)(wsb + 39 * MB);   // 32 MB transposed-weight scratch

  pt_embed<<<(MROWS * D_DIM) / 256, 256, 0, stream>>>(ids, tok, pos, x);

  const dim3 g512(D_DIM / 64, MROWS / 64);
  const dim3 t512(D_DIM / 64, D_DIM / 64);
  for (int l = 0; l < NLAYER; ++l) {
    void* const* L = d_in + 3 + l * 18;
    const float* qw  = (const float*)L[0];  const float* qb  = (const float*)L[1];
    const float* kw  = (const float*)L[2];  const float* kb  = (const float*)L[3];
    const float* vw  = (const float*)L[4];  const float* vb  = (const float*)L[5];
    const float* pw  = (const float*)L[6];  const float* pbi = (const float*)L[7];
    const float* f1w = (const float*)L[8];  const float* f1b = (const float*)L[9];
    const float* f2w = (const float*)L[10]; const float* f2b = (const float*)L[11];
    const float* phase = (const float*)L[12];
    const float* iw    = (const float*)L[13];
    const float* g1 = (const float*)L[14]; const float* b1 = (const float*)L[15];
    const float* g2 = (const float*)L[16]; const float* b2 = (const float*)L[17];

    pt_layernorm<<<MROWS, 256, 0, stream>>>(x, g1, b1, xnb);

    pt_transpose_w<<<t512, 256, 0, stream>>>(qw, wt, D_DIM, D_DIM);
    pt_gemm<<<g512, 128, 0, stream>>>(xnb, wt, qb, nullptr, q, MROWS, D_DIM, D_DIM, 0, 1);
    pt_transpose_w<<<t512, 256, 0, stream>>>(kw, wt, D_DIM, D_DIM);
    pt_gemm<<<g512, 128, 0, stream>>>(xnb, wt, kb, nullptr, k, MROWS, D_DIM, D_DIM, 0, 1);
    pt_transpose_w<<<t512, 256, 0, stream>>>(vw, wt, D_DIM, D_DIM);
    pt_gemm<<<g512, 128, 0, stream>>>(xnb, wt, vb, nullptr, v, MROWS, D_DIM, D_DIM, 0, 1);
    pt_transpose_v<<<dim3(8, 8, 2), 256, 0, stream>>>(v, vt);

    pt_scores<<<dim3(8, 8, 16), 128, 0, stream>>>(q, k, sc);
    for (int wl = 0; wl < NWL; ++wl) {
      pt_softmax<<<16 * S_LEN, 256, 0, stream>>>(sc, phase, P, wl);
      pt_av<<<dim3(1, 8, 16), 128, 0, stream>>>(P, vt, attn, iw, wl, wl != 0);
    }

    pt_pack<<<(MROWS * D_DIM) / 256, 256, 0, stream>>>(attn, attnb);
    pt_transpose_w<<<t512, 256, 0, stream>>>(pw, wt, D_DIM, D_DIM);
    pt_gemm<<<g512, 128, 0, stream>>>(attnb, wt, pbi, x, x, MROWS, D_DIM, D_DIM, 0, 0);

    pt_layernorm<<<MROWS, 256, 0, stream>>>(x, g2, b2, xnb);
    pt_transpose_w<<<dim3(4 * D_DIM / 64, D_DIM / 64), 256, 0, stream>>>(f1w, wt, D_DIM, 4 * D_DIM);
    pt_gemm<<<dim3(4 * D_DIM / 64, MROWS / 64), 128, 0, stream>>>(
        xnb, wt, f1b, nullptr, ffnb, MROWS, 4 * D_DIM, D_DIM, 1, 1);
    pt_transpose_w<<<dim3(D_DIM / 64, 4 * D_DIM / 64), 256, 0, stream>>>(f2w, wt, 4 * D_DIM, D_DIM);
    pt_gemm<<<g512, 128, 0, stream>>>(ffnb, wt, f2b, x, x, MROWS, D_DIM, 4 * D_DIM, 0, 0);
  }

  pt_pack<<<(MROWS * D_DIM) / 256, 256, 0, stream>>>(x, xb);
  pt_transpose_w<<<dim3(32000 / 64, D_DIM / 64), 256, 0, stream>>>(ow_final, wt, D_DIM, 32000);
  pt_gemm<<<dim3(32000 / 64, MROWS / 64), 128, 0, stream>>>(
      xb, wt, ob_final, nullptr, (float*)d_out, MROWS, 32000, D_DIM, 0, 0);
}
